// Time_branch_2d_82334523064445
// MI455X (gfx1250) — compile-verified, matrix operand order checked by reading, not run on hardware
//
#include <hip/hip_runtime.h>
#include <hip/hip_bf16.h>

// Dims from the reference
#define DB    8
#define DR    96
#define DC    64
#define DCIN  8
#define DCOUT 64
#define DW    8
#define DCT   4
#define KFC   (DCOUT * DR)   // 6144  (fc contraction length)
#define MFC   (DB * DC)      // 512   (fc M rows = b*C+c)

typedef __attribute__((ext_vector_type(16))) _Float16 v16h;
typedef __attribute__((ext_vector_type(8)))  float    v8f;
typedef __attribute__((ext_vector_type(4)))  int      v4i;

// ---- CDNA5 async global->LDS path (guarded; falls back to vector copy) ----
#if __has_builtin(__builtin_amdgcn_global_load_async_to_lds_b128)
#define HAS_ASYNC_LDS 1
#else
#define HAS_ASYNC_LDS 0
#endif

#if HAS_ASYNC_LDS
#if __has_builtin(__builtin_amdgcn_s_wait_asynccnt)
#define WAIT_ASYNC() __builtin_amdgcn_s_wait_asynccnt(0)
#else
#define WAIT_ASYNC() asm volatile("s_wait_asynccnt 0x0" ::: "memory")
#endif
#else
#define WAIT_ASYNC() ((void)0)
#endif

__device__ __forceinline__ void copy16_g2lds(const _Float16* gsrc, _Float16* ldst) {
#if HAS_ASYNC_LDS
  __builtin_amdgcn_global_load_async_to_lds_b128(
      (__attribute__((address_space(1))) v4i*)gsrc,
      (__attribute__((address_space(3))) v4i*)ldst, 0, 0);
#else
  *(uint4*)ldst = *(const uint4*)gsrc;
#endif
}

__device__ __forceinline__ v8f wmma_f16(v16h a, v16h b, v8f c) {
  // emits v_wmma_f32_16x16x32_f16
  return __builtin_amdgcn_wmma_f32_16x16x32_f16(false, a, false, b, (short)0, c,
                                                false, false);
}

// A-fragment (16x32 f16) from LDS row-major [M][K], leading dim lda.
// ISA layout: lanes 0-15 row M=lane, VGPR0-3 K=0..7, VGPR4-7 K=16..23;
//             lanes 16-31 same rows, K=8..15 / 24..31.
__device__ __forceinline__ v16h frag_A_h(const _Float16* A, int lda) {
  int lane = threadIdx.x & 31;
  int row  = lane & 15;
  int koff = (lane >> 4) << 3;
  v16h a;
#pragma unroll
  for (int v = 0; v < 8; ++v) {
    int kb = ((v & 4) << 2) + koff + ((v & 3) << 1);
    a[2 * v]     = A[row * lda + kb];
    a[2 * v + 1] = A[row * lda + kb + 1];
  }
  return a;
}

// Same fragment but sourced from fp32 LDS with on-the-fly convert.
__device__ __forceinline__ v16h frag_A_f(const float* A, int lda) {
  int lane = threadIdx.x & 31;
  int row  = lane & 15;
  int koff = (lane >> 4) << 3;
  v16h a;
#pragma unroll
  for (int v = 0; v < 8; ++v) {
    int kb = ((v & 4) << 2) + koff + ((v & 3) << 1);
    a[2 * v]     = (_Float16)A[row * lda + kb];
    a[2 * v + 1] = (_Float16)A[row * lda + kb + 1];
  }
  return a;
}

// B-fragment (32x16 f16) from LDS stored [K][N] row-major, leading dim ldb.
__device__ __forceinline__ v16h frag_B_h(const _Float16* Bm, int ldb) {
  int lane = threadIdx.x & 31;
  int n    = lane & 15;
  int koff = (lane >> 4) << 3;
  v16h b;
#pragma unroll
  for (int v = 0; v < 8; ++v) {
    int kb = ((v & 4) << 2) + koff + ((v & 3) << 1);
    b[2 * v]     = Bm[kb * ldb + n];
    b[2 * v + 1] = Bm[(kb + 1) * ldb + n];
  }
  return b;
}

struct SmemA {
  float xv[DR + DW];            // xv[i] = x[r=i-8] (zero head for causal window)
  float xm[DR + DW][DCT];
  float hidf[DR][DCOUT];        // pre-gate hidden (fp32)
  float sGd[2 * DCOUT];
  float sGm[2 * DCOUT][DCT];
  float sGb[2 * DCOUT];         // bgd+bgm+bgh
  float sbias[DCOUT];           // bd+bm
  union {
    struct { _Float16 A1[DR][64]; _Float16 B1[64][DCOUT]; } p1;  // deal GEMM
    struct { _Float16 Bg[DCOUT][2 * DCOUT]; } p2;                // gate GEMM (Gh^T)
  } u;
};

// One block per (b, c, g): deal conv (as GEMM) + gated unit, store f16 hid.
__global__ __launch_bounds__(128)
void deal_gate_kernel(const float* __restrict__ x, const float* __restrict__ xmark,
                      const float* __restrict__ Wd, const float* __restrict__ bd,
                      const float* __restrict__ Wm, const float* __restrict__ bm,
                      const float* __restrict__ Gd, const float* __restrict__ bgd,
                      const float* __restrict__ Gm, const float* __restrict__ bgm,
                      const float* __restrict__ Gh, const float* __restrict__ bgh,
                      _Float16* __restrict__ hout) {
  __shared__ SmemA sm;
  const int g = blockIdx.x, c = blockIdx.y, b = blockIdx.z;
  const int tid = threadIdx.x;
  const int w = tid >> 5;
  const int lane = tid & 31;

  // ---- stage x / x_mark (with 8-step zero head) ----
  for (int i = tid; i < DR + DW; i += 128) {
    int rr = i - DW;
    float xvv = 0.f, m0 = 0.f, m1 = 0.f, m2 = 0.f, m3 = 0.f;
    if (rr >= 0) {
      size_t base = ((size_t)(b * DR + rr) * DC + c) * DCIN + g;
      xvv = x[base];
      const float* mp = xmark + base * DCT;
      m0 = mp[0]; m1 = mp[1]; m2 = mp[2]; m3 = mp[3];
    }
    sm.xv[i] = xvv;
    sm.xm[i][0] = m0; sm.xm[i][1] = m1; sm.xm[i][2] = m2; sm.xm[i][3] = m3;
  }
  // ---- B1[k][o] = concat(Wd, Wm) along k, zero-padded K 40->64 ----
  for (int e = tid; e < 64 * DCOUT; e += 128) {
    int kk = e >> 6, o = e & 63;
    int go = g * DCOUT + o;
    float v = 0.f;
    if (kk < DW)                 v = Wd[go * DW + kk];
    else if (kk < DW + DCT * DW) v = Wm[go * (DCT * DW) + (kk - DW)];
    sm.u.p1.B1[kk][o] = (_Float16)v;
  }
  { // gate tables (blockDim == 2*DCOUT)
    int gi = g * 2 * DCOUT + tid;
    sm.sGd[tid] = Gd[gi];
    sm.sGb[tid] = bgd[gi] + bgm[gi] + bgh[gi];
    sm.sGm[tid][0] = Gm[gi * DCT + 0];
    sm.sGm[tid][1] = Gm[gi * DCT + 1];
    sm.sGm[tid][2] = Gm[gi * DCT + 2];
    sm.sGm[tid][3] = Gm[gi * DCT + 3];
  }
  if (tid < DCOUT) sm.sbias[tid] = bd[g * DCOUT + tid] + bm[g * DCOUT + tid];
  __syncthreads();

  // ---- A1[r][col]: causal window matrix (cols 0..7 = x, 8..39 = x_mark, pad) ----
  for (int e = tid; e < DR * 64; e += 128) {
    int r = e >> 6, col = e & 63;
    float v = 0.f;
    if (col < DW)                 v = sm.xv[r + col];
    else if (col < DW + DCT * DW) { int cc = col - DW; v = sm.xm[r + (cc & 7)][cc >> 3]; }
    sm.u.p1.A1[r][col] = (_Float16)v;
  }
  __syncthreads();

  // ---- GEMM1: hid[96][64] = A1 x B1 (+ bd+bm) ----
  for (int t = w; t < 24; t += 4) {
    int mt = t >> 2, nt = t & 3;
    v8f acc = {};
#pragma unroll
    for (int kt = 0; kt < 2; ++kt) {
      v16h a  = frag_A_h(&sm.u.p1.A1[mt * 16][kt * 32], 64);
      v16h bb = frag_B_h(&sm.u.p1.B1[kt * 32][nt * 16], DCOUT);
      acc = wmma_f16(a, bb, acc);
    }
    int o  = nt * 16 + (lane & 15);
    int rb = mt * 16 + ((lane >> 4) << 3);
    float bias = sm.sbias[o];
#pragma unroll
    for (int v = 0; v < 8; ++v) sm.hidf[rb + v][o] = acc[v] + bias;
  }
  __syncthreads();

  // ---- Bg[i][o2] = Gh[g][o2][i] (overwrites A1/B1 region) ----
  for (int e = tid; e < DCOUT * 2 * DCOUT; e += 128) {
    int i = e >> 7, o2 = e & 127;
    sm.u.p2.Bg[i][o2] = (_Float16)Gh[((size_t)g * 2 * DCOUT + o2) * DCOUT + i];
  }
  __syncthreads();

  // ---- GEMM2 (gate pre-activations) + blend + transposed store ----
  const size_t wbase = ((size_t)g * MFC + (size_t)(b * DC + c)) * KFC;
  for (int t = w; t < 24; t += 4) {
    int mt = t >> 2, no = t & 3;
    v8f accS = {}, accT = {};
#pragma unroll
    for (int kt = 0; kt < 2; ++kt) {
      v16h a  = frag_A_f(&sm.hidf[mt * 16][kt * 32], DCOUT);
      v16h bs = frag_B_h(&sm.u.p2.Bg[kt * 32][no * 16], 2 * DCOUT);
      v16h bt = frag_B_h(&sm.u.p2.Bg[kt * 32][DCOUT + no * 16], 2 * DCOUT);
      accS = wmma_f16(a, bs, accS);
      accT = wmma_f16(a, bt, accT);
    }
    int o  = no * 16 + (lane & 15);
    int rb = mt * 16 + ((lane >> 4) << 3);
#pragma unroll
    for (int v = 0; v < 8; ++v) {
      int r = rb + v;
      float xs = sm.xv[r + DW];
      float gs = accS[v] + xs * sm.sGd[o] + sm.sGb[o];
      float gt = accT[v] + xs * sm.sGd[DCOUT + o] + sm.sGb[DCOUT + o];
#pragma unroll
      for (int tt = 0; tt < DCT; ++tt) {
        float xmv = sm.xm[r + DW][tt];
        gs += xmv * sm.sGm[o][tt];
        gt += xmv * sm.sGm[DCOUT + o][tt];
      }
      float sg  = 1.0f / (1.0f + __expf(-gs));
      float tg  = tanhf(gt);
      float h   = sm.hidf[r][o];
      float res = h * sg + (1.0f - sg) * tg;
      hout[wbase + (size_t)o * DR + r] = (_Float16)res;   // k = i*96 + r layout
    }
  }
}

// Per-group GEMM: out[bc][g][o] = sum_k hid[g][bc][k] * FwK[g][k][o] + bf
// M=512, N=64, K=6144; block = 64 rows x 64 cols, 4 waves.
// Double-buffered LDS; A tile staged with async global->LDS when available.
__global__ __launch_bounds__(128)
void fc_gemm_kernel(const _Float16* __restrict__ hin, const float* __restrict__ Fw,
                    const float* __restrict__ bfv, float* __restrict__ out) {
  __shared__ __align__(16) _Float16 As[2][64][32];
  __shared__ _Float16 Bs[2][32][64];
  const int g    = blockIdx.y;
  const int m0   = blockIdx.x * 64;
  const int tid  = threadIdx.x;
  const int w    = tid >> 5;
  const int lane = tid & 31;
  const int NK   = KFC / 32;   // 192 K-chunks

  v8f acc[4] = {v8f{}, v8f{}, v8f{}, v8f{}};
  const _Float16* hg = hin + ((size_t)g * MFC + m0) * KFC;
  const float*    Fg = Fw + (size_t)g * DCOUT * DCOUT * DR;

  auto loadA = [&](int kc, int pb) {
#pragma unroll
    for (int vld = 0; vld < 2; ++vld) {
      int e = tid + vld * 128;                 // 256 x 16B copies
      int row = e >> 2, seg = e & 3;
      copy16_g2lds(hg + (size_t)row * KFC + kc * 32 + seg * 8, &As[pb][row][seg * 8]);
    }
  };
  auto loadB = [&](int kc, int pb) {
    int i  = kc / 3;                           // k = i*96 + r; 96 = 3*32
    int r0 = (kc % 3) * 32;
#pragma unroll
    for (int e4 = 0; e4 < 16; ++e4) {
      int e = tid + e4 * 128;
      int kk = e & 31, n = e >> 5;             // consecutive tid -> contiguous r in Fw
      Bs[pb][kk][n] = (_Float16)Fg[(size_t)n * (DCOUT * DR) + (size_t)i * DR + r0 + kk];
    }
  };

  loadA(0, 0);
  loadB(0, 0);
  WAIT_ASYNC();
  __syncthreads();

  for (int kc = 0; kc < NK; ++kc) {
    int pb = kc & 1;
    if (kc + 1 < NK) {
      if (kc + 4 < NK) {   // warm GL2 ahead of the strided fp32 Fw stream
        int in = (kc + 4) / 3, rn = ((kc + 4) % 3) * 32;
        __builtin_prefetch(Fg + (size_t)(tid >> 1) * (DCOUT * DR) +
                               (size_t)in * DR + rn + (tid & 1) * 16, 0, 1);
      }
      loadA(kc + 1, pb ^ 1);
      loadB(kc + 1, pb ^ 1);
    }
    v16h a = frag_A_h(&As[pb][w * 16][0], 32);
#pragma unroll
    for (int nt = 0; nt < 4; ++nt) {
      v16h bb = frag_B_h(&Bs[pb][0][nt * 16], 64);
      acc[nt] = wmma_f16(a, bb, acc[nt]);
    }
    WAIT_ASYNC();
    __syncthreads();
  }

  int nloc = lane & 15;
  int rb   = (lane >> 4) << 3;
#pragma unroll
  for (int nt = 0; nt < 4; ++nt) {
    int o = nt * 16 + nloc;
    float bias = bfv[g * DCOUT + o];
#pragma unroll
    for (int v = 0; v < 8; ++v) {
      int m = m0 + w * 16 + rb + v;                 // m = b*C + c
      out[((size_t)m * DCIN + g) * DCOUT + o] = acc[nt][v] + bias;
    }
  }
}

extern "C" void kernel_launch(void* const* d_in, const int* in_sizes, int n_in,
                              void* d_out, int out_size, void* d_ws, size_t ws_size,
                              hipStream_t stream) {
  (void)in_sizes; (void)n_in; (void)out_size; (void)ws_size;
  const float* x    = (const float*)d_in[0];
  const float* xmk  = (const float*)d_in[1];
  const float* Wd   = (const float*)d_in[2];
  const float* bd   = (const float*)d_in[3];
  const float* Wm   = (const float*)d_in[4];
  const float* bm   = (const float*)d_in[5];
  const float* Gd   = (const float*)d_in[6];
  const float* bgd  = (const float*)d_in[7];
  const float* Gm   = (const float*)d_in[8];
  const float* bgm  = (const float*)d_in[9];
  const float* Gh   = (const float*)d_in[10];
  const float* bgh  = (const float*)d_in[11];
  const float* Fw   = (const float*)d_in[12];
  const float* bfv  = (const float*)d_in[13];
  _Float16* hws = (_Float16*)d_ws;   // 8*512*6144 f16 = 50.3 MB, fully rewritten each call
  float* out = (float*)d_out;

  deal_gate_kernel<<<dim3(DCIN, DC, DB), 128, 0, stream>>>(
      x, xmk, Wd, bd, Wm, bm, Gd, bgd, Gm, bgm, Gh, bgh, hws);
  fc_gemm_kernel<<<dim3(MFC / 64, DCIN), 128, 0, stream>>>(hws, Fw, bfv, out);
}
